// RNN_1331439862192
// MI455X (gfx1250) — matrix operand order repaired and checked by reading, not compiled, not run on hardware
//
#include <hip/hip_runtime.h>
#include <hip/hip_bf16.h>
#include <math.h>

typedef __attribute__((ext_vector_type(16))) _Float16 v16h;
typedef __attribute__((ext_vector_type(8)))  _Float16 v8h;
typedef __attribute__((ext_vector_type(8)))  float    v8f;

#define B_    256
#define N_    512
#define IN_   57
#define HID_  512
#define OUT_  57
#define NODES 513          // N nodes + 1 sentinel (zeros) row per batch
#define SENT  512

// Workspace layout (halves):
//   Bp : packed Wh in WMMA-B fragment order: [ntile(32)][kchunk(16)][lane(32)][16 halves]
//   H  : hidden states, f16, [B][NODES][HID]
#define BP_HALVES       (32 * 16 * 32 * 16)     // 262144 halves = 512 KB
#define H_OFFSET_BYTES  (BP_HALVES * 2)

// Hardware tanh (V_TANH_F32 transcendental on CDNA5) with fallbacks.
__device__ __forceinline__ float fast_tanh(float x) {
#if __has_builtin(__builtin_amdgcn_tanhf)
    return __builtin_amdgcn_tanhf(x);
#elif __has_builtin(__builtin_amdgcn_tanh_f32)
    return __builtin_amdgcn_tanh_f32(x);
#else
    float y;
    asm volatile("v_tanh_f32 %0, %1" : "=v"(y) : "v"(x));
    return y;
#endif
}

// ---------------------------------------------------------------------------
// Prep: pack Wh = W_ih[IN:, :] (512x512 f32) into f16 WMMA-B fragments, and
// zero the sentinel row H[b][512][:].
// B-fragment packing used here (contiguous k, same order as A): lane l holds column (l&15);
// sub = 2v+h in [0,16) holds K = (l>=16 ? 16 : 0) + sub.
// ---------------------------------------------------------------------------
__global__ __launch_bounds__(256)
void prep_kernel(const float* __restrict__ W_ih,
                 _Float16* __restrict__ Bp,
                 _Float16* __restrict__ H)
{
    int idx = blockIdx.x * 256 + threadIdx.x;
    if (idx < BP_HALVES) {
        int sub  =  idx        & 15;
        int lane = (idx >> 4)  & 31;
        int c    = (idx >> 9)  & 15;
        int nt   =  idx >> 13;
        int col  = nt * 16 + (lane & 15);
        int k    = c * 32 + ((lane >> 4) << 4) + sub;     // K row within Wh
        Bp[idx]  = (_Float16)W_ih[(IN_ + k) * HID_ + col];
    }
    if (idx < B_ * HID_) {
        int b = idx >> 9;
        int n = idx & 511;
        H[(long)b * NODES * HID_ + (long)SENT * HID_ + n] = (_Float16)0.f;
    }
}

// ---------------------------------------------------------------------------
// Main recurrent kernel: 16 workgroups x 256 threads (8 waves).
// WG g owns batches [16g, 16g+16) — trees are independent, so the 512-step
// recurrence needs only workgroup barriers.
// Per step: gather children rows -> 4x v_pk_add_f16 + 4x ds_store_b128 into
// A-fragment LDS; barrier; 8 waves x 4 N-tiles x 16 K-chunks of
// v_wmma_f32_16x16x32_f16 (B fragments: i-invariant base + immediate offsets);
// epilogue adds Wx[val]+b_ih, V_TANH_F32, stores f16 row i; fence + barrier.
// ---------------------------------------------------------------------------
__global__ __launch_bounds__(256)
void tree_rnn_kernel(const int*   __restrict__ left,
                     const int*   __restrict__ right,
                     const int*   __restrict__ values,
                     const float* __restrict__ W_ih,
                     const float* __restrict__ b_ih,
                     const _Float16* __restrict__ Bp,
                     _Float16* __restrict__ H)
{
    __shared__ __align__(32) _Float16 Ash[16 * HID_];   // 16 KB: A fragments
    __shared__ int vals_sh[16];

    const int tid  = threadIdx.x;
    const int lane = tid & 31;
    const int wave = tid >> 5;
    const int b0   = blockIdx.x * 16;

    const int  m   = tid & 15;          // batch row produced in phase 1
    const int  c1  = tid >> 4;          // k-chunk produced in phase 1
    const int  kb  = c1 * 32;
    const long batStride = (long)NODES * HID_;

    // ---------- i-invariant hoisting ----------
    // B fragment base pointers: one per owned N-tile; chunk c is +c*1024 B.
    const v16h* btile[4];
#pragma unroll
    for (int t = 0; t < 4; ++t)
        btile[t] = (const v16h*)(Bp + ((long)(wave * 4 + t) * 16 * 32 + lane) * 16);

    // bias per owned output column
    const int ncl = lane & 15;
    const int mro = (lane >> 4) * 8;
    float bias[4];
#pragma unroll
    for (int t = 0; t < 4; ++t)
        bias[t] = b_ih[(wave * 4 + t) * 16 + ncl];

    // phase-1 pointers
    const int* lptr = left   + (b0 + m) * N_;
    const int* rptr = right  + (b0 + m) * N_;
    const int* vptr = values + (b0 + tid) * N_;          // only tid<16 uses it
    const _Float16* gbase = H + (long)(b0 + m) * batStride + kb;  // + l*HID_
    v8h* ab0 = (v8h*)(Ash + c1 * 512 + m * 16);          // lane m   runs
    v8h* ab1 = (v8h*)(Ash + c1 * 512 + (m + 16) * 16);   // lane m+16 runs
    const _Float16* aread = Ash + lane * 16;             // + c*512

    // output row base pointers (per accumulator row M = mro + r2)
    _Float16* hrow[8];
#pragma unroll
    for (int r2 = 0; r2 < 8; ++r2)
        hrow[r2] = H + (long)(b0 + mro + r2) * batStride + ncl;

    for (int i = 0; i < N_; ++i) {
        // ---------------- phase 1: gather + pack A ----------------
        int l = lptr[i]; if (l < 0) l = SENT;
        int r = rptr[i]; if (r < 0) r = SENT;
        if (tid < 16) vals_sh[tid] = vptr[i];

        const v8h* pl = (const v8h*)(gbase + (long)l * HID_);
        const v8h* pr = (const v8h*)(gbase + (long)r * HID_);
        v8h l0 = pl[0], l1 = pl[1], l2 = pl[2], l3 = pl[3];
        v8h r0 = pr[0], r1 = pr[1], r2v = pr[2], r3 = pr[3];
        // A-fragment runs: lane m   <- k[0..8), k[8..16)
        //                  lane m+16<- k[16..24), k[24..32)
        ab0[0] = l0 + r0;      // v_pk_add_f16 x4
        ab0[1] = l1 + r1;
        ab1[0] = l2 + r2v;
        ab1[1] = l3 + r3;
        __syncthreads();

        // ---------------- phase 2: WMMA GEMM ----------------
        v8f acc[4];
#pragma unroll
        for (int t = 0; t < 4; ++t)
            acc[t] = (v8f){0.f, 0.f, 0.f, 0.f, 0.f, 0.f, 0.f, 0.f};

#pragma unroll
        for (int c = 0; c < 16; ++c) {
            v16h af = *(const v16h*)(aread + c * 512);
#pragma unroll
            for (int t = 0; t < 4; ++t)
                acc[t] = __builtin_amdgcn_wmma_f32_16x16x32_f16(
                             false, af, false, btile[t][c * 32],
                             (short)0, acc[t], false, false);
        }

        // ---------------- phase 3: Xp gather + tanh + store ----------------
        const long irow = (long)i * HID_;
#pragma unroll
        for (int t = 0; t < 4; ++t) {
            const int n = (wave * 4 + t) * 16 + ncl;
#pragma unroll
            for (int r2 = 0; r2 < 8; ++r2) {
                float x  = W_ih[vals_sh[mro + r2] * HID_ + n] + bias[t];
                float hv = fast_tanh(acc[t][r2] + x);
                hrow[r2][irow + (wave * 4 + t) * 16] = (_Float16)hv;
            }
        }
        __threadfence();    // make row i visible to all waves' next-step gathers
        __syncthreads();
    }
}

// ---------------------------------------------------------------------------
// Output: logits = H[:, N-1] @ W_o + b_o ; log_softmax over 57 classes.
// One block (64 threads) per batch element.
// ---------------------------------------------------------------------------
__global__ __launch_bounds__(64)
void out_kernel(const _Float16* __restrict__ H,
                const float* __restrict__ W_o,
                const float* __restrict__ b_o,
                float* __restrict__ out)
{
    __shared__ float red[64];
    const int b = blockIdx.x;
    const int o = threadIdx.x;
    const _Float16* h = H + ((long)b * NODES + (N_ - 1)) * HID_;

    float logit = -INFINITY;
    if (o < OUT_) {
        float acc = b_o[o];
        for (int k = 0; k < HID_; ++k)
            acc += (float)h[k] * W_o[k * OUT_ + o];
        logit = acc;
    }

    red[o] = logit;
    __syncthreads();
    for (int s = 32; s > 0; s >>= 1) {
        if (o < s) red[o] = fmaxf(red[o], red[o + s]);
        __syncthreads();
    }
    float mx = red[0];
    __syncthreads();

    red[o] = (o < OUT_) ? expf(logit - mx) : 0.f;
    __syncthreads();
    for (int s = 32; s > 0; s >>= 1) {
        if (o < s) red[o] += red[o + s];
        __syncthreads();
    }
    float lse = logf(red[0]) + mx;
    if (o < OUT_) out[b * OUT_ + o] = logit - lse;
}

// ---------------------------------------------------------------------------
extern "C" void kernel_launch(void* const* d_in, const int* in_sizes, int n_in,
                              void* d_out, int out_size, void* d_ws, size_t ws_size,
                              hipStream_t stream)
{
    const int*   left   = (const int*)  d_in[0];
    const int*   right  = (const int*)  d_in[1];
    const int*   values = (const int*)  d_in[2];
    const float* W_ih   = (const float*)d_in[3];
    const float* b_ih   = (const float*)d_in[4];
    const float* W_o    = (const float*)d_in[5];
    const float* b_o    = (const float*)d_in[6];
    float* out = (float*)d_out;

    _Float16* Bp = (_Float16*)d_ws;
    _Float16* H  = (_Float16*)((char*)d_ws + H_OFFSET_BYTES);

    prep_kernel    <<<(BP_HALVES + 255) / 256, 256, 0, stream>>>(W_ih, Bp, H);
    tree_rnn_kernel<<<B_ / 16, 256, 0, stream>>>(left, right, values, W_ih, b_ih, Bp, H);
    out_kernel     <<<B_, 64, 0, stream>>>(H, W_o, b_o, out);
}
